// FastRNN_17300128268313
// MI455X (gfx1250) — matrix operand order repaired and checked by reading, not compiled, not run on hardware
//
#include <hip/hip_runtime.h>
#include <math.h>

// ---------------------------------------------------------------------------
// FastRNN for MI455X (gfx1250, wave32, WMMA + Tensor Data Mover).
//   wx = x @ W + bias : parallel bf16 WMMA GEMM, x slabs DMAd to LDS by TDM
//                       (pad_enable gives the stride-33 conflict-free layout).
//   scan over T       : 4 independent workgroups (16 batch rows each); per step
//                       16x512x512 bf16 WMMA GEMM; wx tile for step t DMAd to
//                       LDS by TDM overlapping the GEMM; h fed back through LDS
//                       in pre-packed bf16 A-fragment layout.
// ---------------------------------------------------------------------------

typedef __attribute__((ext_vector_type(16))) __bf16       v16bf;
typedef __attribute__((ext_vector_type(8)))  float        v8f;
typedef __attribute__((ext_vector_type(4)))  unsigned int u32x4;
typedef __attribute__((ext_vector_type(8)))  int          i32x8;
typedef __attribute__((ext_vector_type(4)))  int          i32x4;

#define B_ 64
#define T_ 512
#define I_ 256
#define H_ 512

// ---------------------------------------------------------------------------
// TDM 2-D tile load: global (row-major, row_stride in elements, 4B elements)
// -> LDS at byte offset lds_off. Descriptor per CDNA5 ISA
// §8.3/8.4: group0 = {count|flags, lds_addr, global_addr, type=2},
// group1 = {mask/data_size/pad, tensor dims, tile dims, dim0 stride}.
// pad_interval code: 4 -> every 32 DWORDs, 6 -> every 128 DWORDs;
// pad_amount code 0 -> 1 DWORD.
// ---------------------------------------------------------------------------
__device__ __forceinline__ void tdm_load_2d(unsigned lds_off, const void* gptr,
                                            unsigned tile_d0, unsigned tile_d1,
                                            unsigned long long row_stride,
                                            unsigned pad_en, unsigned pad_interval,
                                            unsigned pad_amount) {
  unsigned long long ga = (unsigned long long)gptr;
  u32x4 g0;
  g0[0] = 1u;                                                // count = 1 (valid)
  g0[1] = lds_off;                                           // lds_addr (bytes)
  g0[2] = (unsigned)ga;                                      // global_addr[31:0]
  g0[3] = (unsigned)((ga >> 32) & 0x01FFFFFFu) | (2u << 30); // addr[56:32] | type=2
  const unsigned long long td = 1ull << 20;                  // big dims: no OOB clip
  i32x8 g1;
  g1[0] = (int)((2u << 16) |                    // data_size = 2 -> 4 bytes
                (pad_en << 20) | (pad_interval << 22) | (pad_amount << 25));
  g1[1] = (int)((td & 0xFFFFu) << 16);                                  // dim0 lo16
  g1[2] = (int)(((td >> 16) & 0xFFFFu) | ((td & 0xFFFFu) << 16));       // dim0 hi | dim1 lo
  g1[3] = (int)(((td >> 16) & 0xFFFFu) | (tile_d0 << 16));              // dim1 hi | tile_dim0
  g1[4] = (int)(tile_d1 & 0xFFFFu);                                     // tile_dim1 (tile_dim2=0)
  g1[5] = (int)(row_stride & 0xFFFFFFFFull);                            // dim0_stride lo
  g1[6] = (int)((row_stride >> 32) & 0xFFFFull);                        // dim0_stride hi
  g1[7] = 0;
  i32x4 z4 = {0, 0, 0, 0};
#if defined(__clang_major__) && (__clang_major__ >= 23)
  i32x8 z8 = {0, 0, 0, 0, 0, 0, 0, 0};
  __builtin_amdgcn_tensor_load_to_lds(g0, g1, z4, z4, z8, 0);
#else
  __builtin_amdgcn_tensor_load_to_lds(g0, g1, z4, z4, 0);
#endif
}

// ---------------------------------------------------------------------------
// Pack a row-major f32 [K,N] matrix into bf16 WMMA B-fragment layout.
// Fragment (kt,nt), lane L, element e: K = kt*32 + (L>>4)*8 + e + (e>=8 ? 8 : 0), N = nt*16 + (L&15)
// ---------------------------------------------------------------------------
__global__ void pack_bfrag(const float* __restrict__ src,
                           unsigned short* __restrict__ dst_raw, int K, int N) {
  __bf16* dst = (__bf16*)dst_raw;
  int idx = blockIdx.x * blockDim.x + threadIdx.x;
  int ntiles = N >> 4;
  int total = (K >> 5) * ntiles * 32;
  if (idx >= total) return;
  int lane = idx & 31;
  int tile = idx >> 5;
  int nt = tile % ntiles;
  int kt = tile / ntiles;
  int n = nt * 16 + (lane & 15);
  int kbase = kt * 32 + (lane >> 4) * 8;
  __bf16* d = dst + (size_t)idx * 16;
#pragma unroll
  for (int e = 0; e < 16; ++e)
    d[e] = (__bf16)src[(size_t)(kbase + e + (e >= 8 ? 8 : 0)) * N + n];
}

// ---------------------------------------------------------------------------
// Kernel 1: wx = x @ W + bias.  Block 256 thr (8 waves), tile 128(M) x 64(N).
// x slabs (128x32 f32, padded to stride 33) double-buffered in LDS via TDM.
// ---------------------------------------------------------------------------
#define XS_STRIDE 33
#define XS_BUF    (128 * XS_STRIDE)   // floats per buffer (16896 B)

__global__ __launch_bounds__(256) void wx_gemm(const float* __restrict__ x,
                                               const unsigned short* __restrict__ Wpk_raw,
                                               const float* __restrict__ bias,
                                               float* __restrict__ wx) {
  const __bf16* Wpk = (const __bf16*)Wpk_raw;
  extern __shared__ float xsmem[];              // 2 x 128 x 33 f32
  const int nb = H_ / 64;                       // 8 N-blocks
  int bm = (int)(blockIdx.x / nb) * 128;
  int bn = (int)(blockIdx.x % nb) * 64;
  int tid  = threadIdx.x;
  int wave = tid >> 5;
  int lane = tid & 31;
  int half = lane >> 4;
  int lr   = lane & 15;

  v8f acc[4] = {};

  if (wave == 0)                                // prologue DMA: kt = 0 slab
    tdm_load_2d(0u, x + (size_t)bm * I_, 32, 128, I_, 1, 4, 0);

  for (int kt = 0; kt < I_ / 32; ++kt) {
    if (wave == 0) __builtin_amdgcn_s_wait_tensorcnt(0);
    __syncthreads();                            // slab kt visible to all waves
    if (wave == 0 && kt + 1 < I_ / 32)          // DMA kt+1 overlaps compute kt
      tdm_load_2d((unsigned)(((kt + 1) & 1) * XS_BUF * 4),
                  x + (size_t)bm * I_ + (kt + 1) * 32, 32, 128, I_, 1, 4, 0);

    const float* xs = xsmem + (kt & 1) * XS_BUF;
    // A fragment per ISA 16-bit A 16x32 layout: row = lane&15,
    // pair p<4 -> K = half*8+2p, p>=4 -> K = 16+half*8+2(p-4)
    v16bf afrag;
#pragma unroll
    for (int p = 0; p < 8; ++p) {
      int kb = (p < 4) ? (half * 8 + 2 * p) : (16 + half * 8 + 2 * (p - 4));
      const float* r = xs + (wave * 16 + lr) * XS_STRIDE + kb;
      afrag[2 * p]     = (__bf16)r[0];
      afrag[2 * p + 1] = (__bf16)r[1];
    }
#pragma unroll
    for (int j = 0; j < 4; ++j) {
      int nt = (bn >> 4) + j;
      v16bf bfrag = *(const v16bf*)(Wpk + ((((size_t)kt * 32 + nt) * 32 + lane) << 4));
      acc[j] = __builtin_amdgcn_wmma_f32_16x16x32_bf16(
          false, afrag, false, bfrag, (short)0, acc[j], false, false);
    }
  }
  // C/D layout: lane holds col n=lane&15, rows r + (lane>>4)*8. Fold bias.
#pragma unroll
  for (int j = 0; j < 4; ++j) {
    int n = bn + j * 16 + lr;
    float bv = bias[n];
#pragma unroll
    for (int r = 0; r < 8; ++r) {
      int m = bm + wave * 16 + half * 8 + r;
      wx[(size_t)m * H_ + n] = acc[j][r] + bv;
    }
  }
}

// ---------------------------------------------------------------------------
// Kernel 2: recurrence. One block per 16 batch rows (4 blocks, independent).
// 256 threads = 8 waves; wave w owns output cols [w*64, w*64+64).
// LDS: habuf = h in bf16 A-fragment layout (stride 24 elems = 48B, b128-aligned)
//      wxl   = wx tile for step t, DMAd by TDM (pad every 128 DW -> stride 516)
// ---------------------------------------------------------------------------
#define HA_STRIDE  24
#define HA_BYTES   (16 * 32 * HA_STRIDE * 2)    // 24576 B
#define WXL_STRIDE 516                          // 512 + 4 pad DWORDs per row

__device__ __forceinline__ float fast_tanh(float x) {
  x = fminf(fmaxf(x, -15.f), 15.f);
  float e = __expf(2.f * x);
  return (e - 1.f) / (e + 1.f);
}

__global__ __launch_bounds__(256) void fastrnn_scan(const float* __restrict__ wx,
                                                    const unsigned short* __restrict__ Upk_raw,
                                                    const float* __restrict__ alpha,
                                                    const float* __restrict__ beta,
                                                    float* __restrict__ out) {
  const __bf16* Upk = (const __bf16*)Upk_raw;
  extern __shared__ char smem[];
  __bf16* habuf = (__bf16*)smem;                // offset 0
  float*  wxl   = (float*)(smem + HA_BYTES);    // offset 24576

  int m0   = blockIdx.x * 16;                   // batch base
  int tid  = threadIdx.x;
  int wave = tid >> 5;                          // 0..7
  int lane = tid & 31;
  int half = lane >> 4;
  int lr   = lane & 15;

  for (int i = tid; i < 16 * 32 * HA_STRIDE; i += 256) habuf[i] = (__bf16)0.f;

  float sa = 1.f / (1.f + __expf(-alpha[0]));   // sigmoid(alpha)
  float sb = 1.f / (1.f + __expf(-beta[0]));    // sigmoid(beta)

  float hreg[4][8] = {};                        // h0 = 0 (f32 carried state)
  __syncthreads();

  for (int t = 0; t < T_; ++t) {
    // DMA this step's 16x512 wx tile into LDS; overlaps the GEMM below.
    if (wave == 0)
      tdm_load_2d(HA_BYTES, wx + ((size_t)m0 * T_ + t) * H_,
                  H_, 16, (unsigned long long)T_ * H_, 1, 6, 0);

    // GEMM: acc = h_{t-1} @ U  (16x512x512), 1-deep software pipeline on K.
    v8f acc[4] = {};
    v16bf a_cur = *(const v16bf*)(habuf + (size_t)lane * HA_STRIDE);
    v16bf b_cur[4];
#pragma unroll
    for (int j = 0; j < 4; ++j)
      b_cur[j] = *(const v16bf*)(Upk + (((size_t)(wave * 4 + j)) * 32 + lane) * 16);
#pragma unroll
    for (int kt = 0; kt < 16; ++kt) {
      v16bf a_nxt = a_cur;
      v16bf b_nxt[4] = {b_cur[0], b_cur[1], b_cur[2], b_cur[3]};
      if (kt + 1 < 16) {                        // preload next K-slice fragments
        a_nxt = *(const v16bf*)(habuf + ((size_t)(kt + 1) * 32 + lane) * HA_STRIDE);
#pragma unroll
        for (int j = 0; j < 4; ++j)
          b_nxt[j] = *(const v16bf*)(Upk +
              ((((size_t)(kt + 1) * 32) + (wave * 4 + j)) * 32 + lane) * 16);
      }
#pragma unroll
      for (int j = 0; j < 4; ++j)
        acc[j] = __builtin_amdgcn_wmma_f32_16x16x32_bf16(
            false, a_cur, false, b_cur[j], (short)0, acc[j], false, false);
      a_cur = a_nxt;
#pragma unroll
      for (int j = 0; j < 4; ++j) b_cur[j] = b_nxt[j];
    }

    if (wave == 0) __builtin_amdgcn_s_wait_tensorcnt(0);   // wx tile landed
    __syncthreads();                            // habuf reads + wx tile published

    // Elementwise: h = sb*h + sa*tanh(acc + wx); write out + repack habuf.
#pragma unroll
    for (int j = 0; j < 4; ++j) {
      int n = wave * 64 + j * 16 + lr;          // h column == next-step K index
      int kt = n >> 5;
      int kl = n & 31;
      int upper = kl >> 4;                      // inverse of A-frag K mapping
      int kk = kl & 15;
      int hhalf = kk >> 3;
      int e = 2 * ((upper << 2) + ((kk & 7) >> 1)) + (kk & 1);
#pragma unroll
      for (int r = 0; r < 8; ++r) {
        int m = half * 8 + r;                   // row within 16-batch tile
        float pre  = acc[j][r] + wxl[m * WXL_STRIDE + n + (n >> 7)];
        float c    = fast_tanh(pre);
        float hnew = sb * hreg[j][r] + sa * c;
        hreg[j][r] = hnew;
        out[((size_t)(m0 + m) * T_ + t) * H_ + n] = hnew;
        int lane_t = hhalf * 16 + m;            // A-frag lane holding (m, n)
        habuf[((size_t)kt * 32 + lane_t) * HA_STRIDE + e] = (__bf16)hnew;
      }
    }
    __syncthreads();                            // habuf/wxl ready for step t+1
  }
}

// ---------------------------------------------------------------------------
// Host launcher. ws: [ wx f32 64MB | Wpk bf16 256KB | Upk bf16 512KB ]
// ---------------------------------------------------------------------------
extern "C" void kernel_launch(void* const* d_in, const int* in_sizes, int n_in,
                              void* d_out, int out_size, void* d_ws, size_t ws_size,
                              hipStream_t stream) {
  const float* x     = (const float*)d_in[0];
  const float* W     = (const float*)d_in[1];
  const float* U     = (const float*)d_in[2];
  const float* bias  = (const float*)d_in[3];
  const float* alpha = (const float*)d_in[4];
  const float* beta  = (const float*)d_in[5];
  float* out = (float*)d_out;

  char* ws = (char*)d_ws;
  float* wx = (float*)ws;
  size_t wx_bytes  = (size_t)B_ * T_ * H_ * sizeof(float);            // 64 MB
  unsigned short* Wpk = (unsigned short*)(ws + wx_bytes);             // 256 KB
  size_t wpk_bytes = (size_t)I_ * H_ * sizeof(unsigned short);
  unsigned short* Upk = (unsigned short*)(ws + wx_bytes + wpk_bytes); // 512 KB

  int totalW = (I_ / 32) * (H_ / 16) * 32;
  pack_bfrag<<<(totalW + 255) / 256, 256, 0, stream>>>(W, Wpk, I_, H_);
  int totalU = (H_ / 32) * (H_ / 16) * 32;
  pack_bfrag<<<(totalU + 255) / 256, 256, 0, stream>>>(U, Upk, H_, H_);

  int grid1 = ((B_ * T_) / 128) * (H_ / 64);                          // 2048 blocks
  size_t smem1 = 2 * XS_BUF * sizeof(float);                          // 33792 B
  wx_gemm<<<grid1, 256, smem1, stream>>>(x, Wpk, bias, wx);

  size_t smem2 = HA_BYTES + 16 * WXL_STRIDE * sizeof(float);          // 57600 B
  fastrnn_scan<<<B_ / 16, 256, smem2, stream>>>(wx, Upk, alpha, beta, out);
}
